// StageTransitionBlock_76905684402459
// MI455X (gfx1250) — matrix operand order repaired and checked by reading, not compiled, not run on hardware
//
#include <hip/hip_runtime.h>
#include <math.h>

typedef __attribute__((ext_vector_type(16))) _Float16 v16h;
typedef __attribute__((ext_vector_type(8)))  _Float16 v8h;
typedef __attribute__((ext_vector_type(8)))  float    v8f;
typedef __attribute__((ext_vector_type(4)))  int      v4i;

#define B_     4
#define L_     6272
#define LQ_    1568
#define DIM_   384
#define DOUT_  768
#define HEADS_ 8
#define HD_    96
#define MLP_   3072

// ---------------- async global->LDS copy (gfx1250 path w/ portable fallback) ----------------
#if defined(__has_builtin)
#  if __has_builtin(__builtin_amdgcn_global_load_async_to_lds_b128) && \
      __has_builtin(__builtin_amdgcn_s_wait_asynccnt)
#    define HAVE_ASYNC_LDS 1
#  endif
#endif
#ifndef HAVE_ASYNC_LDS
#  define HAVE_ASYNC_LDS 0
#endif

#if HAVE_ASYNC_LDS
typedef __attribute__((address_space(1))) v4i gv4i;   // global-memory int4
typedef __attribute__((address_space(3))) v4i lv4i;   // LDS int4
#endif

__device__ __forceinline__ void cp_b128(const _Float16* g, _Float16* l) {
#if HAVE_ASYNC_LDS
  __builtin_amdgcn_global_load_async_to_lds_b128((gv4i*)g, (lv4i*)l, 0, 0);
#else
  *(v8h*)l = *(const v8h*)g;
#endif
}
__device__ __forceinline__ void cp_wait_keep1() {   // allow 1 outstanding prefetch
#if HAVE_ASYNC_LDS
  __builtin_amdgcn_s_wait_asynccnt(1);
#endif
}
__device__ __forceinline__ void cp_wait_all() {
#if HAVE_ASYNC_LDS
  __builtin_amdgcn_s_wait_asynccnt(0);
#endif
}

// ---------------- WMMA helpers ----------------
__device__ __forceinline__ v8f wmma_f16(v16h a, v16h b, v8f c) {
  return __builtin_amdgcn_wmma_f32_16x16x32_f16(
      /*neg_a=*/false, a, /*neg_b=*/false, b,
      /*c_mod=*/(short)0, c, /*reuse_a=*/false, /*reuse_b=*/false);
}

__device__ __forceinline__ v16h pack16(v8h lo, v8h hi) {
  v16h r;
#pragma unroll
  for (int i = 0; i < 8; ++i) { r[i] = lo[i]; r[i + 8] = hi[i]; }
  return r;
}

// A-matrix fragment (16x32 f16): lane holds row m=lane&15,
// halves 0..15 -> K = k0+offA+j (offA = 16 for upper lane half)
__device__ __forceinline__ v16h load_a_frag(const _Float16* __restrict__ A,
                                            size_t row, int ldk, int k0, int lane) {
  const int offA = (lane & 16) ? 16 : 0;
  const v8h* p0 = (const v8h*)(A + row * (size_t)ldk + k0 + offA);
  const v8h* p1 = (const v8h*)(A + row * (size_t)ldk + k0 + 8 + offA);
  return pack16(*p0, *p1);
}

// ---------------- prep: two-stream avg + LN(384) -> f16 ----------------
__global__ __launch_bounds__(128) void k_prep(const float* __restrict__ x,
                                              const float* __restrict__ g1,
                                              const float* __restrict__ b1,
                                              _Float16* __restrict__ x2h,
                                              _Float16* __restrict__ xnh) {
  __shared__ float r1[128], r2[128];
  const int row = blockIdx.x;
  const int tid = threadIdx.x;
  const float* xp = x + (size_t)row * (2 * DIM_);
  float v[3];
  float s = 0.f, sq = 0.f;
#pragma unroll
  for (int i = 0; i < 3; ++i) {
    int c = tid + i * 128;
    float a = 0.5f * (xp[c] + xp[c + DIM_]);
    v[i] = a; s += a; sq += a * a;
  }
  r1[tid] = s; r2[tid] = sq; __syncthreads();
  for (int st = 64; st > 0; st >>= 1) {
    if (tid < st) { r1[tid] += r1[tid + st]; r2[tid] += r2[tid + st]; }
    __syncthreads();
  }
  float mean = r1[0] * (1.f / DIM_);
  float var  = r2[0] * (1.f / DIM_) - mean * mean;
  float rstd = rsqrtf(var + 1e-6f);
#pragma unroll
  for (int i = 0; i < 3; ++i) {
    int c = tid + i * 128;
    x2h[(size_t)row * DIM_ + c] = (_Float16)v[i];
    xnh[(size_t)row * DIM_ + c] = (_Float16)((v[i] - mean) * rstd * g1[c] + b1[c]);
  }
}

// ---------------- LN over 768 -> f16 ----------------
__global__ __launch_bounds__(256) void k_ln768(const float* __restrict__ in,
                                               const float* __restrict__ g,
                                               const float* __restrict__ b,
                                               _Float16* __restrict__ out) {
  __shared__ float r1[256], r2[256];
  const int row = blockIdx.x;
  const int tid = threadIdx.x;
  const float* xp = in + (size_t)row * DOUT_;
  float v[3];
  float s = 0.f, sq = 0.f;
#pragma unroll
  for (int i = 0; i < 3; ++i) {
    int c = tid + i * 256;
    float a = xp[c];
    v[i] = a; s += a; sq += a * a;
  }
  r1[tid] = s; r2[tid] = sq; __syncthreads();
  for (int st = 128; st > 0; st >>= 1) {
    if (tid < st) { r1[tid] += r1[tid + st]; r2[tid] += r2[tid + st]; }
    __syncthreads();
  }
  float mean = r1[0] * (1.f / DOUT_);
  float var  = r2[0] * (1.f / DOUT_) - mean * mean;
  float rstd = rsqrtf(var + 1e-6f);
#pragma unroll
  for (int i = 0; i < 3; ++i) {
    int c = tid + i * 256;
    out[(size_t)row * DOUT_ + c] = (_Float16)((v[i] - mean) * rstd * g[c] + b[c]);
  }
}

// ---------------- weight transpose f32[K,N] -> f16[N,K] ----------------
__global__ __launch_bounds__(256) void k_wtrans(const float* __restrict__ w,
                                                _Float16* __restrict__ wt,
                                                int K, int N) {
  int idx = blockIdx.x * 256 + threadIdx.x;
  if (idx >= K * N) return;
  int k = idx / N, n = idx % N;
  wt[(size_t)n * K + k] = (_Float16)w[idx];
}

// ---------------- WMMA GEMM with LDS-staged shared B tile ----------------
// Block = 8 waves covering 128(M) x 64(N); B tile (64x32 f16, 4KB) staged in LDS,
// double-buffered via async global->LDS copies, shared by all 8 waves.
__global__ __launch_bounds__(256) void k_gemm(const _Float16* __restrict__ A,
                                              const _Float16* __restrict__ Bt,
                                              const float* __restrict__ bias,
                                              const float* __restrict__ res,
                                              float* __restrict__ outF,
                                              _Float16* __restrict__ outH,
                                              int M, int N, int K, int do_gelu) {
  __shared__ __align__(32) _Float16 Bs[2][64 * 32];
  const int tid  = threadIdx.x;
  const int lane = tid & 31;
  const int wid  = tid >> 5;
  const int nbn  = N >> 6;
  const int bm   = blockIdx.x / nbn;
  const int bn   = blockIdx.x % nbn;
  const int m0   = (bm * 8 + wid) << 4;
  const int n0   = bn << 6;
  const int lm   = lane & 15;
  const int kbB  = (lane & 16) ? 16 : 0;
  const size_t mrow = (size_t)(m0 + lm);

  // staging assignment: thread -> (column, 8-half segment) of the 64x32 tile
  const int sc = tid >> 2;          // 0..63
  const int sk = (tid & 3) * 8;     // 0,8,16,24
  const _Float16* bsrc = Bt + (size_t)(n0 + sc) * K + sk;
  _Float16* bdst0 = &Bs[0][sc * 32 + sk];
  _Float16* bdst1 = &Bs[1][sc * 32 + sk];

  v8f acc[4];
#pragma unroll
  for (int c = 0; c < 4; ++c)
#pragma unroll
    for (int r = 0; r < 8; ++r) acc[c][r] = 0.f;

  const int nk = K >> 5;
  cp_b128(bsrc, bdst0);             // stage k-slab 0 into buffer 0

  for (int i = 0; i < nk; ++i) {
    const int k0 = i << 5;
    // prefetch next k-slab into the other buffer
    if (i + 1 < nk) {
      cp_b128(bsrc + (size_t)(k0 + 32), (i & 1) ? bdst0 : bdst1);
      cp_wait_keep1();
    } else {
      cp_wait_all();
    }
    __syncthreads();                // staged tile visible to all waves

    v16h af = load_a_frag(A, mrow, K, k0, lane);
    const _Float16* bbuf = Bs[i & 1];
#pragma unroll
    for (int c = 0; c < 4; ++c) {
      v16h bf = *(const v16h*)(bbuf + (c * 16 + lm) * 32 + kbB);
      acc[c] = wmma_f16(af, bf, acc[c]);
    }
    __syncthreads();                // reads done before buffer is re-staged
  }

  const int off = (lane & 16) ? 8 : 0;
#pragma unroll
  for (int c = 0; c < 4; ++c) {
    int col = n0 + c * 16 + lm;
    float bb = bias[col];
#pragma unroll
    for (int r = 0; r < 8; ++r) {
      int row = m0 + r + off;
      float v = acc[c][r] + bb;
      if (do_gelu) v = 0.5f * v * (1.f + erff(v * 0.70710678118654752f));
      size_t oi = (size_t)row * N + col;
      if (res)  v += res[oi];
      if (outF) outF[oi] = v;
      if (outH) outH[oi] = (_Float16)v;
    }
  }
}

// ---------------- depthwise conv3d pool (k=3, s=(1,2,2), pad=1) + LN(96) ----------------
// modes: 0=xr(f32 [B,Lq,768]) 1=q(f32+f16 [BH,Lq,96]) 2=k(f16 [BH,Lq,96]) 3=v(f16 [BH,96,Lq])
__global__ __launch_bounds__(128) void k_pool(const _Float16* __restrict__ in,
                                              int rowStride, int sOff,
                                              const float* __restrict__ pw,
                                              const float* __restrict__ pb,
                                              const float* __restrict__ g,
                                              const float* __restrict__ beta,
                                              float* __restrict__ outF,
                                              _Float16* __restrict__ outH,
                                              int mode) {
  __shared__ float r1[128], r2[128];
  const int d = threadIdx.x;
  const int gid = blockIdx.x;
  const int opos = gid % LQ_;
  const int bh = gid / LQ_;
  const int b = bh >> 3, h = bh & 7;
  const int t = opos / 196, rem = opos % 196;
  const int y = rem / 14, xx = rem % 14;

  float acc = 0.f;
  if (d < HD_) {
    const _Float16* base = in + sOff + h * HD_ + d;
#pragma unroll
    for (int dt = 0; dt < 3; ++dt) {
      int it = t + dt - 1;
      if (it < 0 || it >= 8) continue;
#pragma unroll
      for (int dy = 0; dy < 3; ++dy) {
        int iy = 2 * y + dy - 1;
        if (iy < 0 || iy >= 28) continue;
#pragma unroll
        for (int dx = 0; dx < 3; ++dx) {
          int ix = 2 * xx + dx - 1;
          if (ix < 0 || ix >= 28) continue;
          int l = (it * 28 + iy) * 28 + ix;
          float wv = pw[((dt * 3 + dy) * 3 + dx) * HD_ + d];
          acc += (float)base[((size_t)(b * L_ + l)) * rowStride] * wv;
        }
      }
    }
    acc += pb[d];
  }
  r1[d] = (d < HD_) ? acc : 0.f;
  r2[d] = (d < HD_) ? acc * acc : 0.f;
  __syncthreads();
  for (int st = 64; st > 0; st >>= 1) {
    if (d < st) { r1[d] += r1[d + st]; r2[d] += r2[d + st]; }
    __syncthreads();
  }
  float mean = r1[0] * (1.f / HD_);
  float var  = r2[0] * (1.f / HD_) - mean * mean;
  float rstd = rsqrtf(var + 1e-6f);
  if (d < HD_) {
    float o = (acc - mean) * rstd * g[d] + beta[d];
    if (mode == 0) {
      outF[((size_t)(b * LQ_ + opos)) * DOUT_ + h * HD_ + d] = o;
    } else if (mode == 1) {
      size_t idx = ((size_t)bh * LQ_ + opos) * HD_ + d;
      outF[idx] = o;
      outH[idx] = (_Float16)o;
    } else if (mode == 2) {
      outH[((size_t)bh * LQ_ + opos) * HD_ + d] = (_Float16)o;
    } else {
      outH[((size_t)bh * HD_ + d) * LQ_ + opos] = (_Float16)o;
    }
  }
}

// ---------------- flash attention ----------------
// per wave: one 16-row q tile of one (b,h). q/k: f16 [BH,Lq,96], vT: f16 [BH,96,Lq].
// out: o = softmax(q k^T / sqrt(96)) v + q  -> f16 [B,Lq,768]
__global__ __launch_bounds__(256) void k_flash(const _Float16* __restrict__ qh,
                                               const _Float16* __restrict__ kh,
                                               const _Float16* __restrict__ vT,
                                               const float* __restrict__ qf,
                                               _Float16* __restrict__ outH) {
  __shared__ _Float16 lds[8][16 * 32];
  const int lane = threadIdx.x & 31;
  const int wid  = threadIdx.x >> 5;
  const int tile = blockIdx.x * 8 + wid;         // 0..3135, grid is exact
  const int qt = tile % (LQ_ / 16);
  const int bh = tile / (LQ_ / 16);
  const int b = bh >> 3, h = bh & 7;
  const int q0 = qt * 16;
  const int lm = lane & 15;
  const int offA = (lane & 16) ? 8 : 0;          // also the C-layout row offset
  const int kbB = (lane & 16) ? 16 : 0;

  const _Float16* qbh = qh + (size_t)bh * LQ_ * HD_;
  const _Float16* kbh = kh + (size_t)bh * LQ_ * HD_;
  const _Float16* vbh = vT + (size_t)bh * HD_ * LQ_;

  // preload q A-fragments for K = 0..95 (3 slabs of 32)
  v16h qa[3];
#pragma unroll
  for (int ks = 0; ks < 3; ++ks)
    qa[ks] = load_a_frag(qbh, (size_t)(q0 + lm), HD_, ks * 32, lane);

  v8f o[6];
#pragma unroll
  for (int c = 0; c < 6; ++c)
#pragma unroll
    for (int r = 0; r < 8; ++r) o[c][r] = 0.f;
  float mi[8], li[8];
#pragma unroll
  for (int r = 0; r < 8; ++r) { mi[r] = -1e30f; li[r] = 0.f; }

  const float scale = 0.1020620726159658f;       // 96^-0.5

  for (int kt = 0; kt < LQ_; kt += 32) {
    v8f s0, s1;
#pragma unroll
    for (int r = 0; r < 8; ++r) { s0[r] = 0.f; s1[r] = 0.f; }
#pragma unroll
    for (int ks = 0; ks < 3; ++ks) {
      v16h b0 = *(const v16h*)(kbh + (size_t)(kt + lm) * HD_ + ks * 32 + kbB);
      v16h b1 = *(const v16h*)(kbh + (size_t)(kt + 16 + lm) * HD_ + ks * 32 + kbB);
      s0 = wmma_f16(qa[ks], b0, s0);
      s1 = wmma_f16(qa[ks], b1, s1);
    }

    // online softmax update (row r+offA lives in this lane's 16-lane half)
    float p0[8], p1[8];
#pragma unroll
    for (int r = 0; r < 8; ++r) {
      float a = s0[r] * scale, bb = s1[r] * scale;
      float tmax = fmaxf(a, bb);
#pragma unroll
      for (int mk = 8; mk >= 1; mk >>= 1) tmax = fmaxf(tmax, __shfl_xor(tmax, mk, 32));
      float mnew = fmaxf(mi[r], tmax);
      float alpha = __expf(mi[r] - mnew);
      float e0 = __expf(a - mnew), e1 = __expf(bb - mnew);
      float ps = e0 + e1;
#pragma unroll
      for (int mk = 8; mk >= 1; mk >>= 1) ps += __shfl_xor(ps, mk, 32);
      li[r] = li[r] * alpha + ps;
      mi[r] = mnew;
      p0[r] = e0; p1[r] = e1;
#pragma unroll
      for (int c = 0; c < 6; ++c) o[c][r] *= alpha;
    }

    // stage P (16x32 f16) through LDS to convert C-layout -> A-layout
#pragma unroll
    for (int r = 0; r < 8; ++r) {
      lds[wid][(r + offA) * 32 + lm]      = (_Float16)p0[r];
      lds[wid][(r + offA) * 32 + 16 + lm] = (_Float16)p1[r];
    }
    __syncthreads();
    const v8h* pp0 = (const v8h*)&lds[wid][lm * 32 + kbB];
    const v8h* pp1 = (const v8h*)&lds[wid][lm * 32 + kbB + 8];
    v16h pf = pack16(*pp0, *pp1);
    __syncthreads();

    // O += P x V  (K = 32 keys, N = 96 dims -> 6 WMMAs)
#pragma unroll
    for (int c = 0; c < 6; ++c) {
      v16h vf = *(const v16h*)(vbh + (size_t)(c * 16 + lm) * LQ_ + kt + kbB);
      o[c] = wmma_f16(pf, vf, o[c]);
    }
  }

  // epilogue: normalize, add pooled-q residual, store f16 in [B,Lq,768]
#pragma unroll
  for (int c = 0; c < 6; ++c) {
    int dim = c * 16 + lm;
#pragma unroll
    for (int r = 0; r < 8; ++r) {
      int row = q0 + r + offA;
      float val = o[c][r] / li[r] + qf[((size_t)bh * LQ_ + row) * HD_ + dim];
      outH[((size_t)(b * LQ_ + row)) * DOUT_ + h * HD_ + dim] = (_Float16)val;
    }
  }
}

// ---------------- host-side orchestration ----------------
extern "C" void kernel_launch(void* const* d_in, const int* in_sizes, int n_in,
                              void* d_out, int out_size, void* d_ws, size_t ws_size,
                              hipStream_t stream) {
  (void)in_sizes; (void)n_in; (void)out_size; (void)ws_size;
  const float* x     = (const float*)d_in[0];
  const float* g1    = (const float*)d_in[1];
  const float* b1    = (const float*)d_in[2];
  const float* wqkv  = (const float*)d_in[3];
  const float* bqkv  = (const float*)d_in[4];
  const float* pqw   = (const float*)d_in[5];
  const float* pqb   = (const float*)d_in[6];
  const float* gq    = (const float*)d_in[7];
  const float* bq    = (const float*)d_in[8];
  const float* pkw   = (const float*)d_in[9];
  const float* pkb   = (const float*)d_in[10];
  const float* gk    = (const float*)d_in[11];
  const float* bk    = (const float*)d_in[12];
  const float* pvw   = (const float*)d_in[13];
  const float* pvb   = (const float*)d_in[14];
  const float* gv    = (const float*)d_in[15];
  const float* bv    = (const float*)d_in[16];
  const float* wproj = (const float*)d_in[17];
  const float* bproj = (const float*)d_in[18];
  const float* wres  = (const float*)d_in[19];
  const float* bres  = (const float*)d_in[20];
  const float* g2    = (const float*)d_in[21];
  const float* b2    = (const float*)d_in[22];
  const float* w1    = (const float*)d_in[23];
  const float* bm1   = (const float*)d_in[24];
  const float* w2    = (const float*)d_in[25];
  const float* bm2   = (const float*)d_in[26];
  float* out = (float*)d_out;

  const int R  = B_ * L_;    // 25088 (divisible by 128)
  const int Rq = B_ * LQ_;   // 6272  (divisible by 128)

  char* ws = (char*)d_ws;
  size_t cur = 0;
  auto alloc = [&](size_t bytes) -> char* {
    char* p = ws + cur;
    cur += (bytes + 255) & ~(size_t)255;
    return p;
  };

  _Float16* x2h    = (_Float16*)alloc((size_t)R * DIM_ * 2);
  _Float16* xnh    = (_Float16*)alloc((size_t)R * DIM_ * 2);
  _Float16* wresT  = (_Float16*)alloc((size_t)DOUT_ * DIM_ * 2);
  _Float16* wqkvT  = (_Float16*)alloc((size_t)(3 * DOUT_) * DIM_ * 2);
  _Float16* wprojT = (_Float16*)alloc((size_t)DOUT_ * DOUT_ * 2);
  _Float16* w1T    = (_Float16*)alloc((size_t)MLP_ * DOUT_ * 2);
  _Float16* w2T    = (_Float16*)alloc((size_t)DOUT_ * MLP_ * 2);
  _Float16* xrpreh = (_Float16*)alloc((size_t)R * DOUT_ * 2);
  _Float16* qkvh   = (_Float16*)alloc((size_t)R * 3 * DOUT_ * 2);
  float*    xrpool = (float*)   alloc((size_t)Rq * DOUT_ * 4);
  float*    qf     = (float*)   alloc((size_t)Rq * DOUT_ * 4);   // [BH,Lq,96]
  _Float16* qh2    = (_Float16*)alloc((size_t)Rq * DOUT_ * 2);
  _Float16* kh2    = (_Float16*)alloc((size_t)Rq * DOUT_ * 2);
  _Float16* vTh    = (_Float16*)alloc((size_t)Rq * DOUT_ * 2);
  _Float16* oatth  = (_Float16*)alloc((size_t)Rq * DOUT_ * 2);
  float*    xmid   = (float*)   alloc((size_t)Rq * DOUT_ * 4);
  _Float16* xmh    = (_Float16*)alloc((size_t)Rq * DOUT_ * 2);
  _Float16* h1h    = (_Float16*)alloc((size_t)Rq * MLP_ * 2);

  // 1) two-stream avg + LN1 -> f16 activations
  k_prep<<<R, 128, 0, stream>>>(x, g1, b1, x2h, xnh);

  // 2) weight transposes -> f16 [N,K]
  auto wt = [&](const float* w, _Float16* d, int K, int N) {
    int n = K * N;
    k_wtrans<<<(n + 255) / 256, 256, 0, stream>>>(w, d, K, N);
  };
  wt(wres,  wresT,  DIM_,  DOUT_);
  wt(wqkv,  wqkvT,  DIM_,  3 * DOUT_);
  wt(wproj, wprojT, DOUT_, DOUT_);
  wt(w1,    w1T,    DOUT_, MLP_);
  wt(w2,    w2T,    MLP_,  DOUT_);

  auto gemm = [&](const _Float16* A, const _Float16* Bt, const float* bias,
                  const float* res, float* oF, _Float16* oH,
                  int M, int N, int K, int gelu) {
    int nblk = (M / 128) * (N / 64);
    k_gemm<<<nblk, 256, 0, stream>>>(A, Bt, bias, res, oF, oH, M, N, K, gelu);
  };

  // 3) residual-path linear; 4) QKV projection (both f16 out for pooling)
  gemm(x2h, wresT, bres, nullptr, nullptr, xrpreh, R, DOUT_, DIM_, 0);
  gemm(xnh, wqkvT, bqkv, nullptr, nullptr, qkvh, R, 3 * DOUT_, DIM_, 0);

  // 5) conv3d pools + per-head LN
  const int poolGrid = B_ * HEADS_ * LQ_;
  k_pool<<<poolGrid, 128, 0, stream>>>(xrpreh, DOUT_, 0,         pqw, pqb, gq, bq, xrpool, nullptr, 0);
  k_pool<<<poolGrid, 128, 0, stream>>>(qkvh, 3 * DOUT_, 0,       pqw, pqb, gq, bq, qf, qh2, 1);
  k_pool<<<poolGrid, 128, 0, stream>>>(qkvh, 3 * DOUT_, DOUT_,   pkw, pkb, gk, bk, nullptr, kh2, 2);
  k_pool<<<poolGrid, 128, 0, stream>>>(qkvh, 3 * DOUT_, 2*DOUT_, pvw, pvb, gv, bv, nullptr, vTh, 3);

  // 6) flash attention (+ q residual), f16 out [B,Lq,768]
  const int fTiles = B_ * HEADS_ * (LQ_ / 16);   // 3136, exactly 8 waves/block
  k_flash<<<fTiles / 8, 256, 0, stream>>>(qh2, kh2, vTh, qf, oatth);

  // 7) output projection + xr residual -> x_mid (f32)
  gemm(oatth, wprojT, bproj, xrpool, xmid, nullptr, Rq, DOUT_, DOUT_, 0);

  // 8) LN2 -> f16
  k_ln768<<<Rq, 256, 0, stream>>>(xmid, g2, b2, xmh);

  // 9) MLP up + exact GELU (f16 out); 10) MLP down + x_mid residual -> d_out
  gemm(xmh, w1T, bm1, nullptr, nullptr, h1h, Rq, MLP_, DOUT_, 1);
  gemm(h1h, w2T, bm2, xmid, out, nullptr, Rq, DOUT_, MLP_, 0);
}
